// MultiHeadEncoderDecoderAttention_79465484910962
// MI455X (gfx1250) — matrix operand order repaired and checked by reading, not compiled, not run on hardware
//
#include <hip/hip_runtime.h>
#include <hip/hip_bf16.h>
#include <stdint.h>

// Problem constants (from reference)
#define HIDDEN 1024
#define HEADS  16
#define DHEAD  64
#define NBATCH 2
#define T1     2048
#define T2     2048

// Flash-attention tiling
#define BQ   128   // Q rows per workgroup (8 waves x 16)
#define BK   64    // K/V tile (kv positions per iteration)
#define LDK  72    // padded LDS row stride in bf16 elems (144B, 16B-aligned rows)

// Output-projection GEMM tiling
#define GBM  128
#define GBN  128
#define GBK  32
#define LDG  40    // padded stride (80B rows, 16B-aligned)

typedef __attribute__((ext_vector_type(16))) __bf16 v16bf;
typedef __attribute__((ext_vector_type(8)))  __bf16 v8bf;
typedef __attribute__((ext_vector_type(8)))  float  v8f;

union Frag {
    v16bf v;
    struct Halves { v8bf lo, hi; } h;
};

__device__ __forceinline__ uint16_t f2bfbits(float f) {
    uint32_t u = __builtin_bit_cast(uint32_t, f);
    u += 0x7FFFu + ((u >> 16) & 1u);           // round-to-nearest-even
    return (uint16_t)(u >> 16);
}

__device__ __forceinline__ v8f v8f_zero() {
    v8f z = {0.f, 0.f, 0.f, 0.f, 0.f, 0.f, 0.f, 0.f};
    return z;
}

// ---------------------------------------------------------------------------
// Kernel 0: fp32 -> bf16 conversion of wo_w (1M elements, 4 per thread)
// ---------------------------------------------------------------------------
__global__ __launch_bounds__(256) void cvt_w_kernel(const float* __restrict__ w,
                                                    uint16_t* __restrict__ wbf) {
    int i = (blockIdx.x * 256 + threadIdx.x) * 4;
    float4 f = *(const float4*)(w + i);
    wbf[i + 0] = f2bfbits(f.x);
    wbf[i + 1] = f2bfbits(f.y);
    wbf[i + 2] = f2bfbits(f.z);
    wbf[i + 3] = f2bfbits(f.w);
}

// ---------------------------------------------------------------------------
// Kernel 1: flash attention (online softmax), bf16 WMMA, fp32 accumulation.
// One workgroup = one (batch, head, 128-row Q block). 8 waves, each owns a
// 16-row Q strip held in A-fragment registers.
// Writes context as bf16 in [N, T1, HIDDEN] layout to workspace.
// ---------------------------------------------------------------------------
__global__ __launch_bounds__(256) void flash_attn_kernel(
        const float* __restrict__ q,
        const float* __restrict__ kmat,
        const float* __restrict__ vmat,
        const float* __restrict__ mask,
        uint16_t* __restrict__ ctxbf) {

    __shared__ alignas(16) uint16_t Klds[BK * LDK];        // K tile, [kv][d]
    __shared__ alignas(16) uint16_t Vtlds[DHEAD * LDK];    // V tile transposed, [d][kv]
    __shared__ alignas(16) uint16_t Plds[8 * 16 * LDK];    // per-wave P strip, [row][kv]

    const int tid   = threadIdx.x;
    const int wave  = tid >> 5;
    const int lane  = tid & 31;
    const int lhalf = lane >> 4;   // 0/1: which 16-lane half
    const int l16   = lane & 15;

    const int qblocks = T1 / BQ;                    // 16
    const int wg  = blockIdx.x;                     // 0..511
    const int b   = wg / (HEADS * qblocks);
    const int rem = wg % (HEADS * qblocks);
    const int h   = rem / qblocks;
    const int q0  = (rem % qblocks) * BQ;

    const float* qbase = q    + ((size_t)b * T1 + q0) * HIDDEN + h * DHEAD;
    const float* kbase = kmat + (size_t)b * T2 * HIDDEN + h * DHEAD;
    const float* vbase = vmat + (size_t)b * T2 * HIDDEN + h * DHEAD;
    const float* mbase = mask + ((size_t)b * T2 + q0) * T2;     // mask[b][q][k]
    uint16_t*    obase = ctxbf + ((size_t)b * T1 + q0) * HIDDEN + h * DHEAD;

    // ---- Load Q strip into A fragments (16x32 bf16 layout, two K-chunks) ----
    // A layout: lane M = l16; elem e -> K = c*32 + (e>>3)*16 + lhalf*8 + (e&7)
    const int myrow = wave * 16 + l16;
    Frag qf[2];
#pragma unroll
    for (int c = 0; c < 2; ++c) {
        const float* p0 = qbase + (size_t)myrow * HIDDEN + c * 32 + lhalf * 8;
        float4 a0 = *(const float4*)(p0);
        float4 a1 = *(const float4*)(p0 + 4);
        float4 b0 = *(const float4*)(p0 + 16);
        float4 b1 = *(const float4*)(p0 + 20);
        float t[16] = {a0.x, a0.y, a0.z, a0.w, a1.x, a1.y, a1.z, a1.w,
                       b0.x, b0.y, b0.z, b0.w, b1.x, b1.y, b1.z, b1.w};
#pragma unroll
        for (int e = 0; e < 16; ++e)
            qf[c].v[e] = __builtin_bit_cast(__bf16, f2bfbits(t[e]));
    }

    float run_max[8], run_sum[8];
    v8f ctx[4];
#pragma unroll
    for (int r = 0; r < 8; ++r) { run_max[r] = -1e30f; run_sum[r] = 0.f; }
#pragma unroll
    for (int g = 0; g < 4; ++g) ctx[g] = v8f_zero();

    uint16_t* mylP = &Plds[wave * 16 * LDK];

    for (int kv0 = 0; kv0 < T2; kv0 += BK) {
        // ---- Cooperative stage: K row-major, V transposed, fp32 -> bf16 ----
        {
            const int row = tid >> 2, seg = tid & 3;    // 64 rows x 4 segs of 16
            const float* kp = kbase + (size_t)(kv0 + row) * HIDDEN + seg * 16;
            uint16_t* kd = &Klds[row * LDK + seg * 16];
#pragma unroll
            for (int i = 0; i < 16; i += 4) {
                float4 f = *(const float4*)(kp + i);
                kd[i + 0] = f2bfbits(f.x);
                kd[i + 1] = f2bfbits(f.y);
                kd[i + 2] = f2bfbits(f.z);
                kd[i + 3] = f2bfbits(f.w);
            }
            const float* vp = vbase + (size_t)(kv0 + row) * HIDDEN + seg * 16;
#pragma unroll
            for (int i = 0; i < 16; i += 4) {
                float4 f = *(const float4*)(vp + i);
                Vtlds[(seg * 16 + i + 0) * LDK + row] = f2bfbits(f.x);
                Vtlds[(seg * 16 + i + 1) * LDK + row] = f2bfbits(f.y);
                Vtlds[(seg * 16 + i + 2) * LDK + row] = f2bfbits(f.z);
                Vtlds[(seg * 16 + i + 3) * LDK + row] = f2bfbits(f.w);
            }
        }
        __syncthreads();

        // ---- Scores: S[16 x 64] = Q (16x64) * K^T, 4 N-tiles x 2 K-chunks ----
        v8f acc[4];
#pragma unroll
        for (int g = 0; g < 4; ++g) {
            v8f s = v8f_zero();
#pragma unroll
            for (int c = 0; c < 2; ++c) {
                // B layout: lane N = l16; elem e -> K = (e>>3)*16 + lhalf*8 + (e&7)
                Frag bfr;
                const uint16_t* bp = &Klds[(g * 16 + l16) * LDK + c * 32 + lhalf * 8];
                bfr.h.lo = *(const v8bf*)bp;
                bfr.h.hi = *(const v8bf*)(bp + 16);
                s = __builtin_amdgcn_wmma_f32_16x16x32_bf16(
                        false, qf[c].v, false, bfr.v, (short)0, s, false, false);
            }
            acc[g] = s;
        }

        // ---- Add attention mask (C layout: VGPR r -> row r + lhalf*8) ----
#pragma unroll
        for (int g = 0; g < 4; ++g) {
#pragma unroll
            for (int r = 0; r < 8; ++r) {
                const int qrow = wave * 16 + r + lhalf * 8;
                acc[g][r] += mbase[(size_t)qrow * T2 + kv0 + g * 16 + l16];
            }
        }

        // ---- Online softmax (row reductions across 16 lanes of a half) ----
#pragma unroll
        for (int r = 0; r < 8; ++r) {
            float m = fmaxf(fmaxf(acc[0][r], acc[1][r]), fmaxf(acc[2][r], acc[3][r]));
#pragma unroll
            for (int off = 8; off >= 1; off >>= 1)
                m = fmaxf(m, __shfl_xor(m, off, 32));
            const float nm = fmaxf(run_max[r], m);
            const float sc = __expf(run_max[r] - nm);
            float rs = 0.f;
#pragma unroll
            for (int g = 0; g < 4; ++g) {
                float p = __expf(acc[g][r] - nm);
                acc[g][r] = p;
                rs += p;
            }
#pragma unroll
            for (int off = 8; off >= 1; off >>= 1)
                rs += __shfl_xor(rs, off, 32);
            run_sum[r] = run_sum[r] * sc + rs;
            run_max[r] = nm;
#pragma unroll
            for (int g = 0; g < 4; ++g) ctx[g][r] *= sc;
        }

        // ---- Re-layout P (C layout -> A fragments) via per-wave LDS strip ----
#pragma unroll
        for (int g = 0; g < 4; ++g)
#pragma unroll
            for (int r = 0; r < 8; ++r)
                mylP[(r + lhalf * 8) * LDK + g * 16 + l16] = f2bfbits(acc[g][r]);
        // Same-wave LDS ops are in-order; compiler inserts s_wait_dscnt for reads.
        Frag pf[2];
#pragma unroll
        for (int c = 0; c < 2; ++c) {
            const uint16_t* pp = &mylP[l16 * LDK + c * 32 + lhalf * 8];
            pf[c].h.lo = *(const v8bf*)pp;
            pf[c].h.hi = *(const v8bf*)(pp + 16);
        }

        // ---- Context accumulate: ctx += P (16x64) * V (64x64) ----
#pragma unroll
        for (int g = 0; g < 4; ++g) {
#pragma unroll
            for (int c = 0; c < 2; ++c) {
                Frag vf;
                const uint16_t* vp = &Vtlds[(g * 16 + l16) * LDK + c * 32 + lhalf * 8];
                vf.h.lo = *(const v8bf*)vp;
                vf.h.hi = *(const v8bf*)(vp + 16);
                ctx[g] = __builtin_amdgcn_wmma_f32_16x16x32_bf16(
                        false, pf[c].v, false, vf.v, (short)0, ctx[g], false, false);
            }
        }
        __syncthreads();
    }

    // ---- Normalize and store context (bf16, [N,T1,HIDDEN] layout) ----
#pragma unroll
    for (int r = 0; r < 8; ++r) {
        const float inv = 1.0f / run_sum[r];
        const int qrow = wave * 16 + r + lhalf * 8;
#pragma unroll
        for (int g = 0; g < 4; ++g)
            obase[(size_t)qrow * HIDDEN + g * 16 + l16] = f2bfbits(ctx[g][r] * inv);
    }
}

// ---------------------------------------------------------------------------
// Kernel 2: out = relu(ctx @ W^T + b).  out[m,n] = sum_k ctx[m,k] * W[n,k].
// 128x128 tile per workgroup, BK=32 per step, 8 waves x 8 N-tiles, bf16 WMMA.
// ---------------------------------------------------------------------------
__global__ __launch_bounds__(256) void out_proj_kernel(
        const uint16_t* __restrict__ ctxbf,
        const uint16_t* __restrict__ wbf,
        const float* __restrict__ bias,
        float* __restrict__ out) {

    __shared__ alignas(16) uint16_t Alds[GBM * LDG];
    __shared__ alignas(16) uint16_t Blds[GBN * LDG];

    const int tid   = threadIdx.x;
    const int wave  = tid >> 5;
    const int lane  = tid & 31;
    const int lhalf = lane >> 4;
    const int l16   = lane & 15;
    const int m0 = blockIdx.x * GBM;
    const int n0 = blockIdx.y * GBN;

    v8f acc[8];
#pragma unroll
    for (int j = 0; j < 8; ++j) acc[j] = v8f_zero();

    float bv[8];
#pragma unroll
    for (int j = 0; j < 8; ++j) bv[j] = bias[n0 + j * 16 + l16];

    const int lrow = tid >> 1;   // 0..127
    const int lseg = tid & 1;    // 16 bf16 per segment

    for (int kk = 0; kk < HIDDEN; kk += GBK) {
        // Stage A (context rows) and B (weight rows: B[k][n] = W[n][k])
        {
            const uint16_t* ap = ctxbf + (size_t)(m0 + lrow) * HIDDEN + kk + lseg * 16;
            uint4 d0 = *(const uint4*)ap;
            uint4 d1 = *(const uint4*)(ap + 8);
            uint16_t* as = &Alds[lrow * LDG + lseg * 16];
            *(uint4*)as = d0;
            *(uint4*)(as + 8) = d1;

            const uint16_t* bp = wbf + (size_t)(n0 + lrow) * HIDDEN + kk + lseg * 16;
            uint4 e0 = *(const uint4*)bp;
            uint4 e1 = *(const uint4*)(bp + 8);
            uint16_t* bs = &Blds[lrow * LDG + lseg * 16];
            *(uint4*)bs = e0;
            *(uint4*)(bs + 8) = e1;
        }
        __syncthreads();

        Frag af;
        const uint16_t* ap2 = &Alds[(wave * 16 + l16) * LDG + lhalf * 8];
        af.h.lo = *(const v8bf*)ap2;
        af.h.hi = *(const v8bf*)(ap2 + 16);
#pragma unroll
        for (int j = 0; j < 8; ++j) {
            Frag bf2;
            const uint16_t* bp2 = &Blds[(j * 16 + l16) * LDG + lhalf * 8];
            bf2.h.lo = *(const v8bf*)bp2;
            bf2.h.hi = *(const v8bf*)(bp2 + 16);
            acc[j] = __builtin_amdgcn_wmma_f32_16x16x32_bf16(
                    false, af.v, false, bf2.v, (short)0, acc[j], false, false);
        }
        __syncthreads();
    }

    // Epilogue: bias + ReLU, fp32 store
#pragma unroll
    for (int j = 0; j < 8; ++j) {
#pragma unroll
        for (int r = 0; r < 8; ++r) {
            const int m = m0 + wave * 16 + r + lhalf * 8;
            const float v = acc[j][r] + bv[j];
            out[(size_t)m * HIDDEN + n0 + j * 16 + l16] = fmaxf(v, 0.f);
        }
    }
}

// ---------------------------------------------------------------------------
extern "C" void kernel_launch(void* const* d_in, const int* in_sizes, int n_in,
                              void* d_out, int out_size, void* d_ws, size_t ws_size,
                              hipStream_t stream) {
    const float* q    = (const float*)d_in[0];
    const float* k    = (const float*)d_in[1];
    const float* v    = (const float*)d_in[2];
    const float* mask = (const float*)d_in[3];
    const float* wo_w = (const float*)d_in[4];
    const float* wo_b = (const float*)d_in[5];
    float* out = (float*)d_out;

    // Workspace: [0, 8MB) bf16 context [N*T1, HIDDEN]; [8MB, 10MB) bf16 W
    uint16_t* ctxbf = (uint16_t*)d_ws;
    uint16_t* wbf   = ctxbf + (size_t)NBATCH * T1 * HIDDEN;

    // 1) Convert wo_w to bf16: 1M elems, 4/thread
    cvt_w_kernel<<<HIDDEN * HIDDEN / (256 * 4), 256, 0, stream>>>(wo_w, wbf);

    // 2) Flash attention: N * HEADS * (T1/BQ) = 2*16*16 = 512 workgroups
    flash_attn_kernel<<<NBATCH * HEADS * (T1 / BQ), 256, 0, stream>>>(
            q, k, v, mask, ctxbf);

    // 3) Output projection + bias + ReLU: (4096/128) x (1024/128) tiles
    out_proj_kernel<<<dim3((NBATCH * T1) / GBM, HIDDEN / GBN), 256, 0, stream>>>(
            ctxbf, wbf, wo_b, out);
}
